// SingleHeadSelfAttention_90864328114712
// MI455X (gfx1250) — matrix operand order repaired and checked by reading, not compiled, not run on hardware
//
#include <hip/hip_runtime.h>
#include <hip/hip_bf16.h>
#include <cstdint>

// ---------------------------------------------------------------------------
// Single-head causal self-attention with RoPE, MI455X (gfx1250, wave32).
//
// Staged pipeline: compute-bound (~240 GFLOP bf16) vs ~1 GB of staging
// traffic at 23.3 TB/s -> staging S is cheaper than the 8x score recompute a
// fused kernel with head_dim=1024 would force (O accumulator can't fit).
//
//   K1: [Q|K|V] = X * W^T (bf16 WMMA, fp32 acc) + RoPE on Q,K -> bf16 ws
//   K2: S = Q*K^T / 32, causal mask (-3e38), fp32 ws (upper tiles filled fast)
//   K3: row softmax (one wave per row, online max/sum) -> P bf16 ws
//   K4: O = P*V (causal tile skip), fp32 -> d_out
//
// CDNA5 specifics used:
//   - v_wmma_f32_16x16x32_bf16 (fp32 accum) for all three GEMMs
//   - global_load_async_to_lds_b128 (ASYNCcnt) for the pure-copy LDS fills
//     in K2/K4: bypasses VGPRs, frees VALU for WMMA co-issue
//   - s_wait_asynccnt via inline asm; split-counter waits elsewhere by codegen
//   - global_prefetch (via __builtin_prefetch) on the fp32 convert paths in K1
//   - wave32 shuffles for RoPE pair exchange and softmax reductions
//
// Workspace layout (requires ~480 MiB):
//   [0,96MiB)    Q,K,V bf16 (32 MiB each)
//   [96,352MiB)  S fp32  (4*4096*4096*4)
//   [352,480MiB) P bf16  (4*4096*4096*2)
// ---------------------------------------------------------------------------

typedef __attribute__((ext_vector_type(16))) __bf16 v16bf;
typedef __attribute__((ext_vector_type(8)))  __bf16 v8bf;
typedef __attribute__((ext_vector_type(4)))  __bf16 v4bf;
typedef __attribute__((ext_vector_type(8)))  float  v8f;

#define B_   4
#define S_   4096
#define D_   1024
#define DK_  1024

#define LDA 72     // A-tile LDS row stride (bf16 elems), padded vs bank cycle
#define LDB 136    // B-tile (K-major) LDS row stride
#define LDC 136    // bf16 staging row stride

// round-to-nearest-even f32 -> bf16
__device__ __forceinline__ __bf16 f2bf(float f) {
  union { float f; unsigned u; } x; x.f = f;
  unsigned r = x.u + 0x7FFFu + ((x.u >> 16) & 1u);
  unsigned short h = (unsigned short)(r >> 16);
  return __builtin_bit_cast(__bf16, h);
}

// Async 16B/lane global->LDS copy (GLOBAL_LOAD_ASYNC_TO_LDS_B128, ASYNCcnt).
// LDS dest address = low 32 bits of the generic shared pointer (flat aperture
// rule, ISA 10.2).
__device__ __forceinline__ void async_copy_b128(void* lds, const void* g) {
  unsigned ldsa = (unsigned)(uintptr_t)lds;
  unsigned long long ga = (unsigned long long)(uintptr_t)g;
  asm volatile("global_load_async_to_lds_b128 %0, %1, off"
               :: "v"(ldsa), "v"(ga) : "memory");
}
__device__ __forceinline__ void wait_async0() {
  asm volatile("s_wait_asynccnt 0x0" ::: "memory");
}

// A fragment (16x32 bf16, ISA 7.12.2): lanes 0-15 hold M=lane with K-base 0,
// lanes 16-31 hold M=lane-16 with K-base 8. VGPRs 0..3 = K kb..kb+7 (contig),
// VGPRs 4..7 = K kb+16..kb+23 (contig) -> two 16-byte LDS loads per lane.
__device__ __forceinline__ v16bf load_a_frag(const __bf16* rowp, int kb) {
  union { v16bf v; v8bf h[2]; } u;
  u.h[0] = *(const v8bf*)(rowp + kb);
  u.h[1] = *(const v8bf*)(rowp + kb + 16);
  return u.v;
}

// B fragment (32x16 bf16): lane = N column, elements K kb..kb+7, kb+16..kb+23.
// Source in LDS is K-major, so this is a strided gather (stride LDB) per lane.
__device__ __forceinline__ v16bf load_b_frag(const __bf16* p) {
  v16bf v;
#pragma unroll
  for (int s = 0; s < 8; ++s) {
    v[s] = p[s * LDB];
    v[s + 8] = p[(16 + s) * LDB];
  }
  return v;
}

// ---------------------------------------------------------------------------
// Kernel 1: QKV projection + RoPE.  grid = (DK/128, B*S/128, 3), block = 256
// (fp32->bf16 convert + W transpose force the VGPR fill path here)
// ---------------------------------------------------------------------------
__global__ __launch_bounds__(256) void qkv_rope_kernel(
    const float* __restrict__ X, const float* __restrict__ Wq,
    const float* __restrict__ Wk, const float* __restrict__ Wv,
    const int* __restrict__ tok,
    __bf16* __restrict__ Qo, __bf16* __restrict__ Ko, __bf16* __restrict__ Vo)
{
  __shared__ __align__(16) unsigned char smem[35840];
  __bf16* Xs  = (__bf16*)smem;            // [128][LDA]
  __bf16* WsT = (__bf16*)(smem + 18432);  // [64][LDB], K-major (transposed W)
  __bf16* Cs  = (__bf16*)smem;            // reused after compute: [128][LDC]

  const int t = threadIdx.x;
  const int lane = t & 31, wave = t >> 5;
  const int nt = blockIdx.x, mt = blockIdx.y, pz = blockIdx.z;
  const float* W = (pz == 0) ? Wq : (pz == 1) ? Wk : Wv;
  __bf16* O = (pz == 0) ? Qo : (pz == 1) ? Ko : Vo;

  const int arow = wave * 16 + (lane & 15);
  const int akb  = (lane >> 4) << 3;

  v8f acc[8] = {};

  for (int k0 = 0; k0 < D_; k0 += 64) {
    // X tile 128x64 f32 -> bf16 LDS (row-major)
#pragma unroll
    for (int i = 0; i < 8; ++i) {
      int idx = t + i * 256;
      int r = idx >> 4, c4 = idx & 15;
      float4 xv = *(const float4*)(X + (size_t)(mt * 128 + r) * D_ + k0 + c4 * 4);
      v4bf pk = { f2bf(xv.x), f2bf(xv.y), f2bf(xv.z), f2bf(xv.w) };
      *(v4bf*)(Xs + r * LDA + c4 * 4) = pk;
    }
    // W tile 128(n) x 64(k) f32 -> transposed bf16 LDS: WsT[k][n] = W[n][k]
#pragma unroll
    for (int i = 0; i < 8; ++i) {
      int idx = t + i * 256;
      int n = idx >> 4, c4 = idx & 15;
      float4 wv = *(const float4*)(W + (size_t)(nt * 128 + n) * D_ + k0 + c4 * 4);
      WsT[(c4 * 4 + 0) * LDB + n] = f2bf(wv.x);
      WsT[(c4 * 4 + 1) * LDB + n] = f2bf(wv.y);
      WsT[(c4 * 4 + 2) * LDB + n] = f2bf(wv.z);
      WsT[(c4 * 4 + 3) * LDB + n] = f2bf(wv.w);
    }
    if (k0 + 64 < D_) {  // hint next K-chunk into cache (global_prefetch_b8)
      __builtin_prefetch(X + (size_t)(mt * 128 + (t >> 1)) * D_ + k0 + 64, 0, 1);
      __builtin_prefetch(W + (size_t)(nt * 128 + (t >> 1)) * D_ + k0 + 64, 0, 1);
    }
    __syncthreads();
#pragma unroll
    for (int kk = 0; kk < 64; kk += 32) {
      v16bf a = load_a_frag(Xs + arow * LDA + kk, akb);
      const __bf16* bp = WsT + (kk + akb) * LDB + (lane & 15);
#pragma unroll
      for (int n = 0; n < 8; ++n) {
        v16bf bfr = load_b_frag(bp + n * 16);
        acc[n] = __builtin_amdgcn_wmma_f32_16x16x32_bf16(
            false, a, false, bfr, (short)0, acc[n], false, false);
      }
    }
    __syncthreads();
  }

  // RoPE on C fragments (pair partner = adjacent lane), then bf16 staging.
  const int rbase = wave * 16 + ((lane >> 4) << 3);  // C: VGPR v -> row v / v+8
  const int c0 = lane & 15;
#pragma unroll
  for (int n = 0; n < 8; ++n) {
#pragma unroll
    for (int v = 0; v < 8; ++v) {
      float val = acc[n][v];
      int rloc = rbase + v;
      int cloc = n * 16 + c0;
      if (pz < 2) {  // RoPE only on Q and K
        int grow = mt * 128 + rloc;
        float p = (float)tok[grow & (S_ - 1)];
        int cg = nt * 128 + cloc;
        // theta^(-2i/d) = exp(-(2i/d) * ln(10000))
        float inv = __expf((float)(cg >> 1) * (-2.0f / (float)DK_) * 9.210340371976184f);
        float sn, cs;
        __sincosf(p * inv, &sn, &cs);
        float other = __shfl_xor(val, 1, 32);  // even<->odd partner column
        val = (cg & 1) ? (other * sn + val * cs) : (val * cs - other * sn);
      }
      Cs[rloc * LDC + cloc] = f2bf(val);
    }
  }
  __syncthreads();
#pragma unroll
  for (int i = 0; i < 8; ++i) {  // coalesced 16B stores
    int idx = t + i * 256;
    int r = idx >> 4, c8 = idx & 15;
    v8bf vv = *(const v8bf*)(Cs + r * LDC + c8 * 8);
    *(v8bf*)(O + (size_t)(mt * 128 + r) * DK_ + nt * 128 + c8 * 8) = vv;
  }
}

// ---------------------------------------------------------------------------
// Kernel 2: S = Q K^T / sqrt(dk), causal mask.  grid = (32, 32, 4), block 256
// Q tile fill is a pure copy -> async global->LDS; K tile needs transpose.
// ---------------------------------------------------------------------------
__global__ __launch_bounds__(256) void scores_kernel(
    const __bf16* __restrict__ Q, const __bf16* __restrict__ K,
    float* __restrict__ Sc)
{
  const int kt = blockIdx.x, qt = blockIdx.y, b = blockIdx.z;
  const int t = threadIdx.x;
  float* Sb = Sc + (size_t)b * S_ * S_;
  if (kt > qt) {  // tile fully above diagonal: fill mask value, skip GEMM
    float4 f = make_float4(-3e38f, -3e38f, -3e38f, -3e38f);
#pragma unroll
    for (int i = 0; i < 16; ++i) {
      int idx = t + i * 256;
      int r = idx >> 5, c4 = idx & 31;
      *(float4*)(Sb + (size_t)(qt * 128 + r) * S_ + kt * 128 + c4 * 4) = f;
    }
    return;
  }
  __shared__ __align__(16) unsigned char smem[65536];
  __bf16* Qs  = (__bf16*)smem;            // [128][LDA]
  __bf16* KsT = (__bf16*)(smem + 18432);  // [64][LDB]
  float*  Cs  = (float*)smem;             // reused: [128][128] f32 staging

  const int lane = t & 31, wave = t >> 5;
  const __bf16* Qg = Q + (size_t)b * S_ * DK_;
  const __bf16* Kg = K + (size_t)b * S_ * DK_;
  const int arow = wave * 16 + (lane & 15);
  const int akb  = (lane >> 4) << 3;
  v8f acc[8] = {};

  for (int k0 = 0; k0 < DK_; k0 += 64) {
    // Q tile 128x64 bf16: async copy straight into LDS (no VGPR round-trip)
#pragma unroll
    for (int i = 0; i < 4; ++i) {
      int idx = t + i * 256;
      int r = idx >> 3, c8 = idx & 7;
      async_copy_b128(Qs + r * LDA + c8 * 8,
                      Qg + (size_t)(qt * 128 + r) * DK_ + k0 + c8 * 8);
    }
    // K tile: transpose into K-major LDS (must go through VGPRs)
#pragma unroll
    for (int i = 0; i < 4; ++i) {
      int idx = t + i * 256;
      int n = idx >> 3, c8 = idx & 7;
      v8bf kv = *(const v8bf*)(Kg + (size_t)(kt * 128 + n) * DK_ + k0 + c8 * 8);
#pragma unroll
      for (int j = 0; j < 8; ++j) KsT[(c8 * 8 + j) * LDB + n] = kv[j];
    }
    if (k0 + 64 < DK_) {
      __builtin_prefetch(Kg + (size_t)(kt * 128 + (t >> 1)) * DK_ + k0 + 64, 0, 1);
    }
    wait_async0();   // our async writes to LDS landed
    __syncthreads(); // everyone's writes visible
#pragma unroll
    for (int kk = 0; kk < 64; kk += 32) {
      v16bf a = load_a_frag(Qs + arow * LDA + kk, akb);
      const __bf16* bp = KsT + (kk + akb) * LDB + (lane & 15);
#pragma unroll
      for (int n = 0; n < 8; ++n) {
        v16bf bfr = load_b_frag(bp + n * 16);
        acc[n] = __builtin_amdgcn_wmma_f32_16x16x32_bf16(
            false, a, false, bfr, (short)0, acc[n], false, false);
      }
    }
    __syncthreads();
  }

  const int rbase = wave * 16 + ((lane >> 4) << 3);
  const int c0 = lane & 15;
#pragma unroll
  for (int n = 0; n < 8; ++n)
#pragma unroll
    for (int v = 0; v < 8; ++v) {
      int rloc = rbase + v, cloc = n * 16 + c0;
      float val = acc[n][v] * 0.03125f;  // 1/sqrt(1024)
      if (kt * 128 + cloc > qt * 128 + rloc) val = -3e38f;  // causal
      Cs[rloc * 128 + cloc] = val;
    }
  __syncthreads();
#pragma unroll
  for (int i = 0; i < 16; ++i) {
    int idx = t + i * 256;
    int r = idx >> 5, c4 = idx & 31;
    *(float4*)(Sb + (size_t)(qt * 128 + r) * S_ + kt * 128 + c4 * 4) =
        *(const float4*)(Cs + r * 128 + c4 * 4);
  }
}

// ---------------------------------------------------------------------------
// Kernel 3: row softmax, one wave32 per row.  grid = B*S/8, block 256
// ---------------------------------------------------------------------------
__global__ __launch_bounds__(256) void softmax_kernel(
    const float* __restrict__ Sc, __bf16* __restrict__ P)
{
  const int lane = threadIdx.x & 31;
  const int row = blockIdx.x * 8 + (threadIdx.x >> 5);
  const float* srow = Sc + (size_t)row * S_;
  __bf16* prow = P + (size_t)row * S_;

  float m = -3e38f, l = 0.0f;  // online (max, sum) per lane
  for (int i = lane; i < S_; i += 32) {
    float v = srow[i];
    float mn = fmaxf(m, v);
    l = l * __expf(m - mn) + __expf(v - mn);
    m = mn;
  }
#pragma unroll
  for (int off = 1; off < 32; off <<= 1) {  // wave32 combine
    float mo = __shfl_xor(m, off, 32);
    float lo = __shfl_xor(l, off, 32);
    float mn = fmaxf(m, mo);
    l = l * __expf(m - mn) + lo * __expf(mo - mn);
    m = mn;
  }
  float inv = 1.0f / l;  // l >= 1 (diagonal term), safe
  for (int i = lane; i < S_; i += 32)
    prow[i] = f2bf(__expf(srow[i] - m) * inv);  // masked -> exp(-inf)=0
}

// ---------------------------------------------------------------------------
// Kernel 4: O = P V with causal tile skipping.  grid = (8, 32, 4), block 256
// Both tile fills are pure copies -> fully async global->LDS.
// ---------------------------------------------------------------------------
__global__ __launch_bounds__(256) void pv_kernel(
    const __bf16* __restrict__ P, const __bf16* __restrict__ V,
    float* __restrict__ Out)
{
  __shared__ __align__(16) unsigned char smem[65536];
  __bf16* Ps = (__bf16*)smem;            // [128][LDA]
  __bf16* Vs = (__bf16*)(smem + 18432);  // [64][LDB] (V already K-major)
  float*  Cs = (float*)smem;             // reused: [128][128] f32 staging

  const int t = threadIdx.x;
  const int lane = t & 31, wave = t >> 5;
  const int nt = blockIdx.x, qt = blockIdx.y, b = blockIdx.z;
  const __bf16* Pg = P + (size_t)b * S_ * S_;
  const __bf16* Vg = V + (size_t)b * S_ * DK_;
  const int arow = wave * 16 + (lane & 15);
  const int akb  = (lane >> 4) << 3;
  v8f acc[8] = {};
  const int kend = (qt + 1) * 128;  // keys beyond the diagonal tile are zero

  for (int k0 = 0; k0 < kend; k0 += 64) {
#pragma unroll
    for (int i = 0; i < 4; ++i) {
      int idx = t + i * 256;
      int r = idx >> 3, c8 = idx & 7;
      async_copy_b128(Ps + r * LDA + c8 * 8,
                      Pg + (size_t)(qt * 128 + r) * S_ + k0 + c8 * 8);
    }
#pragma unroll
    for (int i = 0; i < 4; ++i) {  // V tile is already K-major: straight copy
      int idx = t + i * 256;
      int k = idx >> 4, c8 = idx & 15;
      async_copy_b128(Vs + k * LDB + c8 * 8,
                      Vg + (size_t)(k0 + k) * DK_ + nt * 128 + c8 * 8);
    }
    wait_async0();
    __syncthreads();
#pragma unroll
    for (int kk = 0; kk < 64; kk += 32) {
      v16bf a = load_a_frag(Ps + arow * LDA + kk, akb);
      const __bf16* bp = Vs + (kk + akb) * LDB + (lane & 15);
#pragma unroll
      for (int n = 0; n < 8; ++n) {
        v16bf bfr = load_b_frag(bp + n * 16);
        acc[n] = __builtin_amdgcn_wmma_f32_16x16x32_bf16(
            false, a, false, bfr, (short)0, acc[n], false, false);
      }
    }
    __syncthreads();
  }

  const int rbase = wave * 16 + ((lane >> 4) << 3);
  const int c0 = lane & 15;
#pragma unroll
  for (int n = 0; n < 8; ++n)
#pragma unroll
    for (int v = 0; v < 8; ++v)
      Cs[(rbase + v) * 128 + n * 16 + c0] = acc[n][v];
  __syncthreads();
#pragma unroll
  for (int i = 0; i < 16; ++i) {
    int idx = t + i * 256;
    int r = idx >> 5, c4 = idx & 31;
    *(float4*)(Out + (size_t)(b * S_ + qt * 128 + r) * DK_ + nt * 128 + c4 * 4) =
        *(const float4*)(Cs + r * 128 + c4 * 4);
  }
}

// ---------------------------------------------------------------------------
extern "C" void kernel_launch(void* const* d_in, const int* in_sizes, int n_in,
                              void* d_out, int out_size, void* d_ws, size_t ws_size,
                              hipStream_t stream) {
  (void)in_sizes; (void)n_in; (void)out_size; (void)ws_size;
  const float* x  = (const float*)d_in[0];
  const float* Wq = (const float*)d_in[1];
  const float* Wk = (const float*)d_in[2];
  const float* Wv = (const float*)d_in[3];
  const int*  tok = (const int*)d_in[4];

  char* ws = (char*)d_ws;
  const size_t qkvB = (size_t)B_ * S_ * DK_ * 2;   // 32 MiB per tensor
  __bf16* Qb = (__bf16*)(ws);
  __bf16* Kb = (__bf16*)(ws + qkvB);
  __bf16* Vb = (__bf16*)(ws + 2 * qkvB);
  float*  Sc = (float*)(ws + 3 * qkvB);            // 256 MiB
  __bf16* Pb = (__bf16*)(ws + 3 * qkvB + (size_t)B_ * S_ * S_ * 4);  // 128 MiB
  float* Out = (float*)d_out;

  qkv_rope_kernel<<<dim3(DK_ / 128, (B_ * S_) / 128, 3), 256, 0, stream>>>(
      x, Wq, Wk, Wv, tok, Qb, Kb, Vb);
  scores_kernel<<<dim3(S_ / 128, S_ / 128, B_), 256, 0, stream>>>(Qb, Kb, Sc);
  softmax_kernel<<<dim3((B_ * S_) / 8), 256, 0, stream>>>(Sc, Pb);
  pv_kernel<<<dim3(DK_ / 128, S_ / 128, B_), 256, 0, stream>>>(Pb, Vb, Out);
}
